// TensorizedRNN_40450001994316
// MI455X (gfx1250) — compile-verified, matrix operand order checked by reading, not run on hardware
//
#include <hip/hip_runtime.h>

typedef __bf16 bf16_t;
typedef __attribute__((ext_vector_type(16))) __bf16 v16bf;
typedef __attribute__((ext_vector_type(8)))  float  v8f;

#define B_SZ 4096
#define H_SZ 2048
#define KJ_SZ 4096   // 2H, the j-dimension
#define BM 512
#define BN 64
#define KT 32
#define NJ (KJ_SZ / KT)   // 128 K-tiles
#define SPITCH 40    // KT + 8 pad, keeps 16B alignment (80B row pitch)
#define WPITCH 40

#if __has_builtin(__builtin_amdgcn_tensor_load_to_lds) && \
    __has_builtin(__builtin_amdgcn_s_wait_tensorcnt)
#define USE_TDM 1
typedef __attribute__((ext_vector_type(4))) unsigned int v4u_t;
typedef __attribute__((ext_vector_type(8))) int v8i_t;
typedef __attribute__((ext_vector_type(4))) int v4i_t;
#else
#define USE_TDM 0
#endif

struct __align__(8) bf4 { bf16_t x, y, z, w; };
struct Q32 { uint4 a, b; };

static __device__ __forceinline__ v16bf make_v16(uint4 lo, uint4 hi) {
  Q32 q; q.a = lo; q.b = hi;
  return __builtin_bit_cast(v16bf, q);
}

// ---------------- fp32 -> bf16 conversion kernels ----------------
__global__ void convert_bf16(const float* __restrict__ src,
                             bf16_t* __restrict__ dst, long n4) {
  long i = (long)blockIdx.x * blockDim.x + threadIdx.x;
  if (i >= n4) return;
  float4 v = ((const float4*)src)[i];
  bf4 o; o.x = (bf16_t)v.x; o.y = (bf16_t)v.y; o.z = (bf16_t)v.z; o.w = (bf16_t)v.w;
  ((bf4*)dst)[i] = o;
}

// s = concat(h0, h1) along columns, converted to bf16, [4096][4096] row-major
__global__ void build_s(const float* __restrict__ h0, const float* __restrict__ h1,
                        bf16_t* __restrict__ dst) {
  long i = (long)blockIdx.x * blockDim.x + threadIdx.x;  // one thread per 4 elems
  long j = i * 4;
  long row = j >> 12;          // / 4096
  long c   = j & 4095;
  const float* src = (c < 2048) ? (h0 + row * 2048 + c)
                                : (h1 + row * 2048 + (c - 2048));
  float4 v = *(const float4*)src;
  bf4 o; o.x = (bf16_t)v.x; o.y = (bf16_t)v.y; o.z = (bf16_t)v.z; o.w = (bf16_t)v.w;
  *(bf4*)(dst + j) = o;
}

// ---------------- merge GEMM: gm = s @ Wmerge (fp32 out) ----------------
// 256 threads = 8 waves (4 along M, 2 along N); block tile 128x64, wave 32x32.
__global__ __launch_bounds__(256, 1)
void merge_gemm(const bf16_t* __restrict__ sb, const bf16_t* __restrict__ Wmb,
                float* __restrict__ gm) {
  __shared__ __align__(16) bf16_t sT[128][SPITCH];   // 10240 B
  __shared__ __align__(16) bf16_t wTm[64][WPITCH];   //  5120 B

  const int tid   = threadIdx.x;
  const int lane  = tid & 31;
  const int wave  = tid >> 5;
  const int waveM = wave & 3;     // 0..3
  const int waveN = wave >> 2;    // 0..1
  const int l15   = lane & 15;
  const int lhalf = lane >> 4;
  const int khalf = lhalf * 8;

  const int bm0 = blockIdx.x * 128;
  const int n0  = blockIdx.y * 64;

  const int wk  = tid >> 3;   // 0..31
  const int wsg = tid & 7;    // 0..7

  v8f acc[2][2];
  v8f vz = {0.f, 0.f, 0.f, 0.f, 0.f, 0.f, 0.f, 0.f};
  acc[0][0] = vz; acc[0][1] = vz; acc[1][0] = vz; acc[1][1] = vz;

#pragma unroll 1
  for (int j0 = 0; j0 < KJ_SZ; j0 += KT) {
    __syncthreads();
#pragma unroll
    for (int it = 0; it < 2; ++it) {          // stage s tile [128 x 32]
      int idx = tid + it * 256;
      int row = idx >> 2;
      int seg = idx & 3;
      uint4 v = *(const uint4*)(sb + (long)(bm0 + row) * KJ_SZ + j0 + seg * 8);
      *(uint4*)&sT[row][seg * 8] = v;
    }
    {                                          // stage Wm tile transposed [n][k]
      uint4 v = *(const uint4*)(Wmb + ((long)j0 + wk) * H_SZ + n0 + wsg * 8);
      const bf16_t* p = (const bf16_t*)&v;
#pragma unroll
      for (int e = 0; e < 8; ++e) wTm[wsg * 8 + e][wk] = p[e];
    }
    __syncthreads();
#pragma unroll
    for (int nt = 0; nt < 2; ++nt) {
      const bf16_t* pb = &wTm[waveN * 32 + nt * 16 + l15][khalf];
      v16bf bfr = make_v16(*(const uint4*)pb, *(const uint4*)(pb + 16));
#pragma unroll
      for (int mt = 0; mt < 2; ++mt) {
        const bf16_t* pa = &sT[waveM * 32 + mt * 16 + l15][khalf];
        v16bf a = make_v16(*(const uint4*)pa, *(const uint4*)(pa + 16));
        acc[mt][nt] = __builtin_amdgcn_wmma_f32_16x16x32_bf16(
            false, a, false, bfr, (short)0, acc[mt][nt], false, false);
      }
    }
  }
#pragma unroll
  for (int mt = 0; mt < 2; ++mt)
#pragma unroll
    for (int nt = 0; nt < 2; ++nt)
#pragma unroll
      for (int r = 0; r < 8; ++r) {
        long grow = (long)bm0 + waveM * 32 + mt * 16 + lhalf * 8 + r;
        int  col  = n0 + waveN * 32 + nt * 16 + l15;
        gm[grow * H_SZ + col] = acc[mt][nt][r];
      }
}

// ---------------- fused bilinear-GRU kernel ----------------
// acc1 = sum_i (x_i * s) @ W1_i ; acc2 = same with W2 ; gm precomputed.
// out  = sigmoid(acc2+b2) * tanh(acc1+b1) + (1-sigmoid)*gm   (written twice)
__global__ __launch_bounds__(512, 1)
void trnn_fused(const bf16_t* __restrict__ W1b, const bf16_t* __restrict__ W2b,
                const bf16_t* __restrict__ sb, const float* __restrict__ gm,
                const float* __restrict__ x0, const float* __restrict__ x1,
                const float* __restrict__ b1, const float* __restrict__ b2,
                float* __restrict__ out) {
  __shared__ __align__(16) bf16_t sT[2][BM][SPITCH];       // 81920 B (dbl buf, TDM)
  __shared__ __align__(16) bf16_t wT[2][2][BN][WPITCH];    // 20480 B (dbl buf)

  const int tid   = threadIdx.x;
  const int lane  = tid & 31;
  const int wave  = tid >> 5;
  const int waveM = wave & 3;          // 4 waves along M
  const int waveN = wave >> 2;         // 4 waves along N
  const int l15   = lane & 15;
  const int lhalf = lane >> 4;         // 0/1
  const int khalf = lhalf * 8;

  const int bm0  = blockIdx.x * BM;
  const int n0   = blockIdx.y * BN;
  const int ncol = n0 + waveN * 16 + l15;       // this lane's output column
  const int mrowbase = waveM * 128 + l15;       // + mt*16 -> block-local row

  // W staging role of this thread: 1 x uint4 (8 bf16) per stage
  const int wmat = tid >> 8;    // 0 -> W1, 1 -> W2
  const int wr   = tid & 255;
  const int wk   = wr >> 3;     // 0..31  (k within tile)
  const int wseg = wr & 7;      // 0..7   (8-col group)

#if USE_TDM
  // Issue a TDM 2D tile load: [BM rows x KT cols] bf16, global row stride KJ_SZ,
  // LDS pitch = SPITCH via pad (every 16 DWORDs insert 4 DWORDs).
  auto issue_tdm = [&](int bufsel, int j0) {
    const unsigned lds = (unsigned)(size_t)&sT[bufsel][0][0];
    const unsigned long long ga =
        (unsigned long long)(size_t)(sb + (long)bm0 * KJ_SZ + j0);
    v4u_t g0;
    g0[0] = 1u;                                   // count=1, user descriptor
    g0[1] = lds;                                  // LDS byte address
    g0[2] = (unsigned)(ga & 0xFFFFFFFFu);         // global_addr[31:0]
    g0[3] = (unsigned)((ga >> 32) & 0x01FFFFFFu)  // global_addr[56:32]
            | (2u << 30);                         // type = 2 (image)
    v8i_t g1;
    // data_size=1 (2B) | pad_enable | pad_interval=3 (16 DW) | pad_amount=3 (4 DW)
    g1[0] = (int)((1u << 16) | (1u << 20) | (3u << 22) | (3u << 25));
    const unsigned td0 = 0x40000000u, td1 = 0x40000000u;  // huge tensor dims
    g1[1] = (int)((td0 & 0xFFFFu) << 16);                       // dim0[15:0]
    g1[2] = (int)(((td0 >> 16) & 0xFFFFu) | ((td1 & 0xFFFFu) << 16));
    g1[3] = (int)(((td1 >> 16) & 0xFFFFu) | ((unsigned)KT << 16)); // tile_dim0=32
    g1[4] = (int)(unsigned)BM;                    // tile_dim1=512, tile_dim2=0
    g1[5] = (int)(unsigned)KJ_SZ;                 // dim0_stride = 4096 elems
    g1[6] = 0;
    g1[7] = 0;
    v4i_t gz; gz[0] = 0; gz[1] = 0; gz[2] = 0; gz[3] = 0;
    v8i_t gz8;
#pragma unroll
    for (int e = 0; e < 8; ++e) gz8[e] = 0;
    __builtin_amdgcn_tensor_load_to_lds(g0, g1, gz, gz, gz8, 0);
  };
#endif

  v8f acc1[8], acc2[8];
  v8f vz = {0.f, 0.f, 0.f, 0.f, 0.f, 0.f, 0.f, 0.f};
#pragma unroll
  for (int mt = 0; mt < 8; ++mt) { acc1[mt] = vz; acc2[mt] = vz; }

#if USE_TDM
  if (wave == 0) issue_tdm(0, 0);       // prologue: first s tile in flight
#endif

#pragma unroll 1
  for (int jj = 0; jj < NJ; ++jj) {
    const int j0   = jj * KT;
    const int scur = jj & 1;
#if USE_TDM
    if (wave == 0) __builtin_amdgcn_s_wait_tensorcnt(0);  // tile(jj) landed
#endif
    __syncthreads();   // publish sT[scur]; prior readers of sT[1-scur]/wT[0] done
#if USE_TDM
    if (wave == 0 && jj + 1 < NJ) issue_tdm(1 - scur, j0 + KT);  // async next tile
#else
    // ---- stage s tile [BM x KT] synchronously ----
#pragma unroll
    for (int it = 0; it < 4; ++it) {
      int idx = tid + it * 512;
      int row = idx >> 2;
      int seg = idx & 3;
      uint4 v = *(const uint4*)(sb + (long)(bm0 + row) * KJ_SZ + j0 + seg * 8);
      *(uint4*)&sT[scur][row][seg * 8] = v;
    }
#endif
    // ---- stage W(i=0) tiles into buffer 0 (transposed [n][k]) ----
    {
      uint4 v = *(const uint4*)((wmat ? W2b : W1b) + ((long)j0 + wk) * H_SZ + n0 + wseg * 8);
      const bf16_t* p = (const bf16_t*)&v;
#pragma unroll
      for (int e = 0; e < 8; ++e) wT[0][wmat][wseg * 8 + e][wk] = p[e];
    }
    __syncthreads();

#pragma unroll 1
    for (int i = 0; i < 8; ++i) {
      // fetch next W tile to registers (none needed after i==7)
      uint4 wnext = uint4{0u, 0u, 0u, 0u};
      const bool wvalid = (i < 7);
      if (wvalid) {
        const bf16_t* wb = wmat ? W2b : W1b;
        wnext = *(const uint4*)(wb + ((long)(i + 1) * KJ_SZ + j0 + wk) * H_SZ + n0 + wseg * 8);
      }
      // per-row x scales (x is tiny and L2-hot)
      const float* xp = (i < 4) ? (x0 + i) : (x1 + (i - 4));
      bf16_t xsb[8];
#pragma unroll
      for (int mt = 0; mt < 8; ++mt)
        xsb[mt] = (bf16_t)xp[(long)(bm0 + mrowbase + mt * 16) * 4];

      const int buf = i & 1;
      // B fragments (W1 slice, W2 slice)
      const bf16_t* pb0 = &wT[buf][0][waveN * 16 + l15][khalf];
      v16bf bfr0 = make_v16(*(const uint4*)pb0, *(const uint4*)(pb0 + 16));
      const bf16_t* pb1 = &wT[buf][1][waveN * 16 + l15][khalf];
      v16bf bfr1 = make_v16(*(const uint4*)pb1, *(const uint4*)(pb1 + 16));
#pragma unroll
      for (int mt = 0; mt < 8; ++mt) {
        const bf16_t* pa = &sT[scur][mrowbase + mt * 16][khalf];
        v16bf a = make_v16(*(const uint4*)pa, *(const uint4*)(pa + 16));
        a = a * xsb[mt];  // per-lane scalar: each lane's A fragment is one M-row
        acc1[mt] = __builtin_amdgcn_wmma_f32_16x16x32_bf16(
            false, a, false, bfr0, (short)0, acc1[mt], false, false);
        acc2[mt] = __builtin_amdgcn_wmma_f32_16x16x32_bf16(
            false, a, false, bfr1, (short)0, acc2[mt], false, false);
      }
      // commit next W tile into the other buffer
      if (wvalid) {
        const bf16_t* p = (const bf16_t*)&wnext;
        const int nbuf = (i + 1) & 1;
#pragma unroll
        for (int e = 0; e < 8; ++e) wT[nbuf][wmat][wseg * 8 + e][wk] = p[e];
      }
      __syncthreads();
    }
  }

  // ---- gate epilogue + dual store (tuple returns new_state twice) ----
  const float bias1 = b1[ncol];
  const float bias2 = b2[ncol];
  const long BH = (long)B_SZ * H_SZ;
#pragma unroll
  for (int mt = 0; mt < 8; ++mt) {
#pragma unroll
    for (int r = 0; r < 8; ++r) {
      long grow = (long)bm0 + waveM * 128 + mt * 16 + lhalf * 8 + r;
      float g1 = acc1[mt][r] + bias1;
      float st = tanhf(g1);
      float g2 = acc2[mt][r] + bias2;
      float u  = 1.0f / (1.0f + __expf(-g2));
      float o  = u * st + (1.0f - u) * gm[grow * H_SZ + ncol];
      out[grow * H_SZ + ncol]      = o;
      out[BH + grow * H_SZ + ncol] = o;
    }
  }
}

extern "C" void kernel_launch(void* const* d_in, const int* in_sizes, int n_in,
                              void* d_out, int out_size, void* d_ws, size_t ws_size,
                              hipStream_t stream) {
  const float* x0 = (const float*)d_in[0];
  const float* x1 = (const float*)d_in[1];
  const float* h0 = (const float*)d_in[2];
  const float* h1 = (const float*)d_in[3];
  const float* W1 = (const float*)d_in[4];
  const float* b1 = (const float*)d_in[5];
  const float* W2 = (const float*)d_in[6];
  const float* b2 = (const float*)d_in[7];
  const float* Wm = (const float*)d_in[8];
  float* out = (float*)d_out;

  const size_t nW = (size_t)8 * KJ_SZ * H_SZ;   // 67,108,864 elems per W tensor
  const size_t nS = (size_t)B_SZ * KJ_SZ;       // 16,777,216
  const size_t nM = (size_t)KJ_SZ * H_SZ;       //  8,388,608

  bf16_t* W1b = (bf16_t*)d_ws;
  bf16_t* W2b = W1b + nW;
  bf16_t* sbf = W2b + nW;
  bf16_t* Wmb = sbf + nS;
  float*  gm  = (float*)(Wmb + nM);             // [B][H] fp32 merge result

  convert_bf16<<<(unsigned)(nW / 4 / 256), 256, 0, stream>>>(W1, W1b, (long)(nW / 4));
  convert_bf16<<<(unsigned)(nW / 4 / 256), 256, 0, stream>>>(W2, W2b, (long)(nW / 4));
  convert_bf16<<<(unsigned)(nM / 4 / 256), 256, 0, stream>>>(Wm, Wmb, (long)(nM / 4));
  build_s<<<(unsigned)(nS / 4 / 256), 256, 0, stream>>>(h0, h1, sbf);

  dim3 mgrid(B_SZ / 128, H_SZ / 64);            // 32 x 32
  merge_gemm<<<mgrid, 256, 0, stream>>>(sbf, Wmb, gm);

  dim3 grid(B_SZ / BM, H_SZ / BN);              // 8 x 32 blocks, 512 threads
  trnn_fused<<<grid, 512, 0, stream>>>(W1b, W2b, sbf, gm, x0, x1, b1, b2, out);
}
